// NemotronHTopkRouter_12360915878721
// MI455X (gfx1250) — matrix operand order repaired and checked by reading, not compiled, not run on hardware
//
#include <hip/hip_runtime.h>

// ---------------------------------------------------------------------------
// NemotronH / DeepSeek-style top-k router for MI455X (gfx1250, wave32, WMMA)
//
//   logits = hs[T,H] @ W[E,H]^T     (fp16 inputs, f32 accum via v_wmma)
//   scores = sigmoid(logits); choice = scores + bias
//   group (8 x 32) top-2 sums -> top-4 groups -> masked top-8 -> norm * 2.5
//
// Roofline: 68.7 GFLOP fp16 matmul vs ~290 MB of HBM traffic (~12.5us @
// 23.3 TB/s) -> GEMM-bound, runs on the WMMA pipe. W (2 MB) and the 32 MB
// logits scratch are L2-resident (192 MB L2).
//
// W k-chunks ([256 x 32] fp16 tiles, row stride 4096) are DMA'd into a
// double-buffered LDS region by the Tensor Data Mover (tensor_load_to_lds),
// with hardware row padding (16 B per 64 B row) to spread LDS banks.
// ---------------------------------------------------------------------------

typedef __attribute__((ext_vector_type(16))) _Float16 v16h;
typedef __attribute__((ext_vector_type(8)))  _Float16 v8h;
typedef __attribute__((ext_vector_type(8)))  float    v8f;
typedef __attribute__((ext_vector_type(4)))  unsigned int u32x4;
typedef __attribute__((ext_vector_type(8)))  int          i32x8;
typedef __attribute__((ext_vector_type(4)))  int          i32x4;

#define T_TOKENS   32768
#define HIDDEN     4096
#define NEXP       256
#define TOPK       8
#define NGROUP     8
#define EG         32     // experts per group
#define TOPKG      4
#define SCALING    2.5f

#define BM         128    // tokens per block (8 waves x 16 tokens)
#define LDS_STRIDE 40     // halves per expert row in LDS (32 + 8 pad halves)

// ---------------------------------------------------------------------------
// TDM: issue a tensor_load_to_lds of one W k-chunk: 2D tile [256 rows x 32
// fp16], tensor row stride = HIDDEN, with LDS padding of 4 DWORDs after
// every 16 DWORDs stored (=> LDS row pitch of 80 B = LDS_STRIDE halves).
// D# layout per CDNA5 ISA ch.8 (group0: count/lds/global/type,
// group1: data_size/pad/dims/tile/strides). Groups 2-3 zero (2D tensor).
// ---------------------------------------------------------------------------
__device__ __forceinline__ void tdm_load_w_chunk(const _Float16* gsrc,
                                                 unsigned lds_off) {
    const unsigned long long ga = (unsigned long long)(uintptr_t)gsrc;

    u32x4 g0;
    g0[0] = 1u;                                   // count=1 (valid user D#)
    g0[1] = lds_off;                              // lds_addr (bytes)
    g0[2] = (unsigned)(ga & 0xFFFFFFFFu);         // global_addr[31:0]
    g0[3] = (unsigned)((ga >> 32) & 0x01FFFFFFu)  // global_addr[56:32]
          | (2u << 30);                           // type=2 ("image")

    i32x8 g1;
    g1[0] = (int)((1u << 16)      // data_size=1 -> 2-byte elements
                | (1u << 20)      // pad_enable
                | (3u << 22)      // pad_interval=3 -> every 16 DWORDs (64 B)
                | (3u << 25));    // pad_amount=3  -> +4 DWORDs (16 B)
    g1[1] = (int)(((unsigned)HIDDEN & 0xFFFFu) << 16);      // tensor_dim0 lo16
    g1[2] = (int)(((unsigned)HIDDEN >> 16)                  // tensor_dim0 hi16
                | (((unsigned)NEXP & 0xFFFFu) << 16));      // tensor_dim1 lo16
    g1[3] = (int)(((unsigned)NEXP >> 16)                    // tensor_dim1 hi16
                | (32u << 16));                             // tile_dim0 = 32
    g1[4] = (int)(256u);                                    // tile_dim1 = 256
    g1[5] = (int)((unsigned)HIDDEN);                        // dim0_stride lo32
    g1[6] = 0;                                              // stride hi / dim1_stride
    g1[7] = 0;

    i32x4 z4 = {0, 0, 0, 0};
#if defined(__clang_major__) && (__clang_major__ >= 23)
    i32x8 z8 = {0, 0, 0, 0, 0, 0, 0, 0};
    __builtin_amdgcn_tensor_load_to_lds(g0, g1, z4, z4, z8, 0);
#else
    __builtin_amdgcn_tensor_load_to_lds(g0, g1, z4, z4, 0);
#endif
}

// ---------------------------------------------------------------------------
// Kernel 1: router GEMM. Block = 256 threads = 8 wave32.
// Wave w computes tokens [blk*128 + w*16, +16) x all 256 experts:
// 16 WMMA C-tiles (16x16 f32) = 128 accumulator VGPRs per lane.
// Wave 0 drives the TDM double buffer; everyone computes.
// ---------------------------------------------------------------------------
__global__ __launch_bounds__(256) void router_gemm_kernel(
    const _Float16* __restrict__ hs,
    const _Float16* __restrict__ wgt,
    float* __restrict__ logits)
{
    __shared__ _Float16 wbuf[2][NEXP * LDS_STRIDE];   // 2 x 20 KB

    const int tid   = threadIdx.x;
    const int wave  = tid >> 5;
    const int lane  = tid & 31;
    const int lhalf = lane >> 4;     // 0 or 1 (which half-wave)
    const int l16   = lane & 15;

    // Opaque never-taken store: the TDM builtin is the only writer of wbuf,
    // keep the optimizer from assuming the LDS is never written.
    if (blockDim.y == 2u) wbuf[0][0] = (_Float16)0.0f;

    const int token_base = blockIdx.x * BM + wave * 16;

    v8f acc[16] = {};                // 16 N-tiles of f32 C, zero-init

    const unsigned ldsoff[2] = {
        (unsigned)(uintptr_t)&wbuf[0][0],
        (unsigned)(uintptr_t)&wbuf[1][0]
    };
    const bool issuer = (wave == 0);

    // Prologue: DMA chunk 0 into buffer 0.
    if (issuer) {
        tdm_load_w_chunk(wgt, ldsoff[0]);
        __builtin_amdgcn_s_wait_tensorcnt(0);
    }
    __syncthreads();

    // A row for this lane (per 16-bit A layout: lanes 0-15 are M=0..15)
    const _Float16* arow = hs + (size_t)(token_base + l16) * HIDDEN;

    int cur = 0;
    for (int kc = 0; kc < HIDDEN; kc += 32) {
        // ---- kick off DMA of the next W chunk into the other buffer
        if (issuer && (kc + 32 < HIDDEN))
            tdm_load_w_chunk(wgt + (kc + 32), ldsoff[cur ^ 1]);

        // hint the A stream forward (global_prefetch_b8)
        __builtin_prefetch(arow + kc + 512, 0, 1);

        // ---- A fragment, 16x32 fp16 (ISA layout):
        // lane<16 : v0-3 = K 0..7,  v4-7 = K 16..23
        // lane>=16: v0-3 = K 8..15, v4-7 = K 24..31
        const int koff = kc + lhalf * 8;
        v8h alo = *(const v8h*)(arow + koff);
        v8h ahi = *(const v8h*)(arow + koff + 16);
        v16h a;
        #pragma unroll
        for (int i = 0; i < 8; ++i) { a[i] = alo[i]; a[i + 8] = ahi[i]; }

        // ---- 16 N-tiles, B fragments rotated one tile ahead so ds_load
        // latency overlaps the previous WMMA.
        // B layout (matches A): lanes 0-15 hold K 0..7,16..23 (col n=l16), lanes 16-31 K 8..15,24..31
        const _Float16* bbase = &wbuf[cur][l16 * LDS_STRIDE + lhalf * 8];
        v8h blo = *(const v8h*)(bbase);
        v8h bhi = *(const v8h*)(bbase + 16);
        #pragma unroll
        for (int nt = 0; nt < 16; ++nt) {
            v16h b;
            #pragma unroll
            for (int i = 0; i < 8; ++i) { b[i] = blo[i]; b[i + 8] = bhi[i]; }
            if (nt < 15) {
                const _Float16* bnext = bbase + (nt + 1) * 16 * LDS_STRIDE;
                blo = *(const v8h*)(bnext);
                bhi = *(const v8h*)(bnext + 16);
            }
            acc[nt] = __builtin_amdgcn_wmma_f32_16x16x32_f16(
                /*neg_a=*/false, a, /*neg_b=*/false, b,
                /*c_mod=*/(short)0, acc[nt],
                /*reuse_a=*/false, /*reuse_b=*/false);
        }

        // ---- retire the in-flight DMA, publish buffer to all waves
        if (issuer) __builtin_amdgcn_s_wait_tensorcnt(0);
        __syncthreads();
        cur ^= 1;
    }

    // ---- epilogue: write f32 logits (rounded through fp16 to mimic the
    // fp16 einsum output that the reference then casts to f32)
    // C layout: VGPR r -> M=r (lanes 0-15) / M=r+8 (lanes 16-31); N = l16
    #pragma unroll
    for (int nt = 0; nt < 16; ++nt) {
        const int e = nt * 16 + l16;
        #pragma unroll
        for (int r = 0; r < 8; ++r) {
            const int tok = token_base + r + lhalf * 8;
            const float v = (float)(_Float16)acc[nt][r];
            logits[(size_t)tok * NEXP + e] = v;
        }
    }
}

// ---------------------------------------------------------------------------
// Kernel 2: per-token routing decision. One thread per token; logits row is
// L2-resident (just written). Bias cached in LDS.
// ---------------------------------------------------------------------------
__global__ __launch_bounds__(256) void router_topk_kernel(
    const float* __restrict__ logits,
    const _Float16* __restrict__ bias,
    int* __restrict__ out_idx,
    float* __restrict__ out_w)
{
    __shared__ float sbias[NEXP];
    if (threadIdx.x < NEXP) sbias[threadIdx.x] = (float)bias[threadIdx.x];
    __syncthreads();

    const int t = blockIdx.x * blockDim.x + threadIdx.x;
    if (t >= T_TOKENS) return;
    const float* row = logits + (size_t)t * NEXP;

    // group_scores[g] = sum of top-2 (sigmoid(logit) + bias) within group
    float gscore[NGROUP];
    #pragma unroll
    for (int g = 0; g < NGROUP; ++g) {
        float m1 = -INFINITY, m2 = -INFINITY;
        for (int j = 0; j < EG; ++j) {
            const int e = g * EG + j;
            const float s = 1.0f / (1.0f + __expf(-row[e])) + sbias[e];
            if (s > m1)      { m2 = m1; m1 = s; }
            else if (s > m2) { m2 = s; }
        }
        gscore[g] = m1 + m2;
    }

    // top-4 groups (strict '>' over ascending g == lax.top_k tie-break)
    unsigned gmask = 0u;
    #pragma unroll
    for (int i = 0; i < TOPKG; ++i) {
        float best = -INFINITY; int bg = 0;
        #pragma unroll
        for (int g = 0; g < NGROUP; ++g)
            if (!((gmask >> g) & 1u) && gscore[g] > best) { best = gscore[g]; bg = g; }
        gmask |= 1u << bg;
    }

    // top-8 over masked scores; masked-out value is exactly 0.0f
    // (matches jnp.where(mask, scores_for_choice, 0.0)). Stable insertion:
    // strict '>' keeps the earlier index on ties, matching lax.top_k.
    float key[TOPK]; int idx[TOPK];
    #pragma unroll
    for (int i = 0; i < TOPK; ++i) { key[i] = -INFINITY; idx[i] = 0; }
    for (int e = 0; e < NEXP; ++e) {
        const int g = e >> 5;
        const float s = ((gmask >> g) & 1u)
                      ? (1.0f / (1.0f + __expf(-row[e])) + sbias[e])
                      : 0.0f;
        if (s > key[TOPK - 1]) {
            int p = TOPK - 1;
            while (p > 0 && s > key[p - 1]) {
                key[p] = key[p - 1]; idx[p] = idx[p - 1]; --p;
            }
            key[p] = s; idx[p] = e;
        }
    }

    // weights = sigmoid scores (NOT +bias) at selected experts, normalized
    float w[TOPK]; float sum = 0.0f;
    #pragma unroll
    for (int i = 0; i < TOPK; ++i) {
        const float s = 1.0f / (1.0f + __expf(-row[idx[i]]));
        w[i] = s; sum += s;
    }
    const float scale = SCALING / (sum + 1e-20f);
    #pragma unroll
    for (int i = 0; i < TOPK; ++i) {
        out_idx[(size_t)t * TOPK + i] = idx[i];
        out_w[(size_t)t * TOPK + i]   = w[i] * scale;
    }
}

// ---------------------------------------------------------------------------
// Launcher. d_ws holds the [T, 256] f32 logits scratch (32 MB).
// d_out: first T*8 elements = topk indices (int32 bits), next T*8 = weights.
// ---------------------------------------------------------------------------
extern "C" void kernel_launch(void* const* d_in, const int* in_sizes, int n_in,
                              void* d_out, int out_size, void* d_ws, size_t ws_size,
                              hipStream_t stream) {
    const _Float16* hs   = (const _Float16*)d_in[0];   // [T, H] fp16
    const _Float16* wgt  = (const _Float16*)d_in[1];   // [E, H] fp16
    const _Float16* bias = (const _Float16*)d_in[2];   // [E]    fp16

    float* logits = (float*)d_ws;                       // [T, E] f32 scratch
    int*   out_idx = (int*)d_out;                       // [T, 8] int32
    float* out_w   = (float*)d_out + (size_t)T_TOKENS * TOPK;  // [T, 8] f32

    router_gemm_kernel<<<T_TOKENS / BM, 256, 0, stream>>>(hs, wgt, logits);
    router_topk_kernel<<<(T_TOKENS + 255) / 256, 256, 0, stream>>>(
        logits, bias, out_idx, out_w);
}
